// GenerativeClassifier_55164559950486
// MI455X (gfx1250) — matrix operand order repaired and checked by reading, not compiled, not run on hardware
//
#include <hip/hip_runtime.h>
#include <stdint.h>

typedef __attribute__((ext_vector_type(16))) __bf16 v16bf;
typedef __attribute__((ext_vector_type(8)))  float  v8f;

#define B_N 8192
#define C_N 1000
#define CP  1024
#define D_N 2048

#define LDSTRIDE 40                 // ushorts per LDS row (32 data + 8 pad), 16B aligned rows
#define TILE (128 * LDSTRIDE)       // ushorts per tile
#define NCHUNK (D_N / 32)

struct Frag32B { uint4 q0, q1; };   // 32 bytes == v16bf

__device__ inline unsigned short f32_to_bf16_rne(float x) {
  union { float f; unsigned int u; } v; v.f = x;
  unsigned int u = v.u;
  u += 0x7fffu + ((u >> 16) & 1u);
  return (unsigned short)(u >> 16);
}
__device__ inline float bf16_to_f32(unsigned short h) {
  union { unsigned int u; float f; } v; v.u = ((unsigned int)h) << 16;
  return v.f;
}
__device__ inline v8f wmma_bf16(v16bf a, v16bf b, v8f c) {
  // D = A(16x32 bf16) * B(32x16 bf16) + C(16x16 f32)
  return __builtin_amdgcn_wmma_f32_16x16x32_bf16(
      /*neg_a=*/false, a, /*neg_b=*/false, b,
      /*c_mod=*/(short)0, c, /*reuse_a=*/false, /*reuse_b=*/false);
}

// low 32 bits of a generic pointer to an LDS object == LDS byte address
__device__ inline unsigned lds_addr32(const void* p) {
  return (unsigned)(unsigned long long)p;
}
// async 16B copy: global memory -> LDS (ASYNCcnt-tracked, no VGPR data path)
__device__ inline void async_copy16(unsigned lds_byte_addr, const void* gptr) {
  unsigned long long ga = (unsigned long long)gptr;
  asm volatile("global_load_async_to_lds_b128 %0, %1, off"
               :: "v"(lds_byte_addr), "v"(ga)
               : "memory");
}
__device__ inline void wait_async_le8() {
  asm volatile("s_wait_asynccnt 0x8" ::: "memory");
}
__device__ inline void wait_async_0() {
  asm volatile("s_wait_asynccnt 0x0" ::: "memory");
}

// ---------------- Kernel 0: split z into bf16 hi/lo, compute z2[b] ----------------
__global__ void k_prep_z(const float* __restrict__ z,
                         unsigned short* __restrict__ zh,
                         unsigned short* __restrict__ zl,
                         float* __restrict__ z2) {
  int b = blockIdx.x, t = threadIdx.x;
  const float* row = z + (size_t)b * D_N;
  float ssq = 0.f;
#pragma unroll
  for (int i = 0; i < D_N / 256; ++i) {
    int d = t + i * 256;
    float x = row[d];
    unsigned short h = f32_to_bf16_rne(x);
    float xl = x - bf16_to_f32(h);
    zh[(size_t)b * D_N + d] = h;
    zl[(size_t)b * D_N + d] = f32_to_bf16_rne(xl);
    ssq += x * x;
  }
  __shared__ float red[256];
  red[t] = ssq; __syncthreads();
  for (int s = 128; s > 0; s >>= 1) { if (t < s) red[t] += red[t + s]; __syncthreads(); }
  if (t == 0) z2[b] = red[0];
}

// ---------------- Kernel 1: same for mu, padded to CP rows ----------------
__global__ void k_prep_mu(const float* __restrict__ mu,
                          unsigned short* __restrict__ mh,
                          unsigned short* __restrict__ ml,
                          float* __restrict__ mu2) {
  int c = blockIdx.x, t = threadIdx.x;
  float ssq = 0.f;
#pragma unroll
  for (int i = 0; i < D_N / 256; ++i) {
    int d = t + i * 256;
    float x = (c < C_N) ? mu[(size_t)c * D_N + d] : 0.0f;
    unsigned short h = f32_to_bf16_rne(x);
    float xl = x - bf16_to_f32(h);
    mh[(size_t)c * D_N + d] = h;
    ml[(size_t)c * D_N + d] = f32_to_bf16_rne(xl);
    ssq += x * x;
  }
  __shared__ float red[256];
  red[t] = ssq; __syncthreads();
  for (int s = 128; s > 0; s >>= 1) { if (t < s) red[t] += red[t + s]; __syncthreads(); }
  if (t == 0) mu2[c] = red[0];
}

// ---------------- Kernel 2: WMMA GEMM (async double-buffered) ----------------
// zz = z2 + mu2 - 2*(z . mu), split-bf16: acc += zl*mh + zh*ml + zh*mh
__global__ __launch_bounds__(256) void k_gemm_zz(
    const unsigned short* __restrict__ zh, const unsigned short* __restrict__ zl,
    const unsigned short* __restrict__ mh, const unsigned short* __restrict__ ml,
    const float* __restrict__ z2, const float* __restrict__ mu2,
    float* __restrict__ zz) {
  extern __shared__ unsigned short smem[];  // 2 buffers x 4 tiles x TILE ushorts = 80KB

  const int n0 = blockIdx.x * 128;
  const int m0 = blockIdx.y * 128;
  const int t = threadIdx.x;
  const int lane = t & 31, w = t >> 5;
  const int wm = w & 3, wn = w >> 2;        // 4 waves in M, 2 in N
  const int half = lane >> 4, mrow = lane & 15;

  v8f acc[2][4];
#pragma unroll
  for (int i = 0; i < 2; ++i)
#pragma unroll
    for (int j = 0; j < 4; ++j) acc[i][j] = (v8f)0.0f;

  const int ldRow = t >> 1;          // 0..127
  const int ldSeg = (t & 1) * 16;    // 0 or 16 (ushorts)

  // LDS byte addresses of this thread's 16B destinations (per tile), per buffer
  const unsigned thrOff = 2u * (unsigned)(ldRow * LDSTRIDE + ldSeg);
  const unsigned bufBytes = 4u * TILE * 2u;
  const unsigned tileBytes = (unsigned)TILE * 2u;
  const unsigned ldsBase = lds_addr32(smem);

  // issue one K-chunk's async loads into buffer `buf` (8 async ops per wave)
  auto issue_chunk = [&](int buf, int k0) {
    size_t zi = (size_t)(m0 + ldRow) * D_N + k0 + ldSeg;
    size_t ci = (size_t)(n0 + ldRow) * D_N + k0 + ldSeg;
    unsigned dst = ldsBase + (unsigned)buf * bufBytes + thrOff;
    async_copy16(dst,                     zh + zi);
    async_copy16(dst + 16,                zh + zi + 8);
    async_copy16(dst + tileBytes,         zl + zi);
    async_copy16(dst + tileBytes + 16,    zl + zi + 8);
    async_copy16(dst + 2 * tileBytes,     mh + ci);
    async_copy16(dst + 2 * tileBytes + 16, mh + ci + 8);
    async_copy16(dst + 3 * tileBytes,     ml + ci);
    async_copy16(dst + 3 * tileBytes + 16, ml + ci + 8);
  };

  issue_chunk(0, 0);

  for (int kc = 0; kc < NCHUNK; ++kc) {
    const int cur = kc & 1;
    if (kc + 1 < NCHUNK) {
      issue_chunk(cur ^ 1, (kc + 1) * 32);
      wait_async_le8();   // my chunk-kc loads landed (in-order completion)
    } else {
      wait_async_0();
    }
    __syncthreads();      // everyone's chunk-kc data visible in LDS

    const unsigned short* sZh = smem + (size_t)cur * 4 * TILE;
    const unsigned short* sZl = sZh + TILE;
    const unsigned short* sMh = sZl + TILE;
    const unsigned short* sMl = sMh + TILE;

    // A fragments (16x32, 16-bit layout): lane<16 -> K{0..7,16..23}, lane>=16 -> +8
    v16bf aHi[2], aLo[2];
#pragma unroll
    for (int mt = 0; mt < 2; ++mt) {
      int r = wm * 32 + mt * 16 + mrow;
      Frag32B fa;
      fa.q0 = *(const uint4*)&sZh[r * LDSTRIDE + 8 * half];
      fa.q1 = *(const uint4*)&sZh[r * LDSTRIDE + 16 + 8 * half];
      aHi[mt] = __builtin_bit_cast(v16bf, fa);
      fa.q0 = *(const uint4*)&sZl[r * LDSTRIDE + 8 * half];
      fa.q1 = *(const uint4*)&sZl[r * LDSTRIDE + 16 + 8 * half];
      aLo[mt] = __builtin_bit_cast(v16bf, fa);
    }
    // B fragments (32x16): lane<16 -> K{0..7,16..23} of column N=lane&15; lane>=16 -> +8
#pragma unroll
    for (int nt = 0; nt < 4; ++nt) {
      int r = wn * 64 + nt * 16 + mrow;
      Frag32B fb;
      fb.q0 = *(const uint4*)&sMh[r * LDSTRIDE + 8 * half];
      fb.q1 = *(const uint4*)&sMh[r * LDSTRIDE + 16 + 8 * half];
      v16bf bHi = __builtin_bit_cast(v16bf, fb);
      fb.q0 = *(const uint4*)&sMl[r * LDSTRIDE + 8 * half];
      fb.q1 = *(const uint4*)&sMl[r * LDSTRIDE + 16 + 8 * half];
      v16bf bLo = __builtin_bit_cast(v16bf, fb);
#pragma unroll
      for (int mt = 0; mt < 2; ++mt) {
        acc[mt][nt] = wmma_bf16(aLo[mt], bHi, acc[mt][nt]);
        acc[mt][nt] = wmma_bf16(aHi[mt], bLo, acc[mt][nt]);
        acc[mt][nt] = wmma_bf16(aHi[mt], bHi, acc[mt][nt]);
      }
    }
    __syncthreads();      // all waves done reading buffer `cur` before it is rewritten
  }

  // Epilogue: C/D layout -> lane: N = lane&15, VGPR v: M = v + 8*(lane>>4)
#pragma unroll
  for (int mt = 0; mt < 2; ++mt) {
#pragma unroll
    for (int nt = 0; nt < 4; ++nt) {
      int gc = n0 + wn * 64 + nt * 16 + mrow;
      float m2 = mu2[gc];
#pragma unroll
      for (int v = 0; v < 8; ++v) {
        int gm = m0 + wm * 32 + mt * 16 + v + 8 * half;
        float val = z2[gm] + m2 - 2.0f * acc[mt][nt][v];
        zz[(size_t)gm * CP + gc] = val;
      }
    }
  }
}

// ---------------- Kernel 3: per-row logsumexp / sums ----------------
__global__ void k_row_reduce(const float* __restrict__ zz, const float* __restrict__ y,
                             float* __restrict__ rowLse, float* __restrict__ rowSum,
                             float* __restrict__ rowZzy) {
  int b = blockIdx.x, t = threadIdx.x;
  const float* row = zz + (size_t)b * CP;
  const float* yrow = y + (size_t)b * C_N;
  float vals[4];
  float lmax = -3.4e38f;
#pragma unroll
  for (int i = 0; i < 4; ++i) {
    int c = t + i * 256;
    float v = (c < C_N) ? (-0.5f * row[c]) : -3.4e38f;
    vals[i] = v;
    lmax = fmaxf(lmax, v);
  }
  __shared__ float red[256];
  red[t] = lmax; __syncthreads();
  for (int s = 128; s > 0; s >>= 1) { if (t < s) red[t] = fmaxf(red[t], red[t + s]); __syncthreads(); }
  float M = red[0];
  __syncthreads();

  float se = 0.f, sl = 0.f, szy = 0.f;
#pragma unroll
  for (int i = 0; i < 4; ++i) {
    int c = t + i * 256;
    if (c < C_N) {
      float v = vals[i];
      se += expf(v - M);
      sl += v;                         // sum of logits (-0.5*zz)
      szy += (-2.0f * v) * yrow[c];    // sum of zz*y
    }
  }
  red[t] = se; __syncthreads();
  for (int s = 128; s > 0; s >>= 1) { if (t < s) red[t] += red[t + s]; __syncthreads(); }
  float SE = red[0]; __syncthreads();
  red[t] = sl; __syncthreads();
  for (int s = 128; s > 0; s >>= 1) { if (t < s) red[t] += red[t + s]; __syncthreads(); }
  float SL = red[0]; __syncthreads();
  red[t] = szy; __syncthreads();
  for (int s = 128; s > 0; s >>= 1) { if (t < s) red[t] += red[t + s]; __syncthreads(); }
  if (t == 0) {
    rowLse[b] = M + logf(SE);  // logsumexp(-0.5*zz) over valid classes
    rowSum[b] = SL;
    rowZzy[b] = red[0];
  }
}

// ---------------- Kernel 4: final means ----------------
__global__ void k_final(const float* __restrict__ rowLse, const float* __restrict__ rowSum,
                        const float* __restrict__ rowZzy, const float* __restrict__ jac,
                        float* __restrict__ out) {
  int t = threadIdx.x;
  const float logC = logf((float)C_N);
  const float invD = 1.0f / (float)D_N;
  float sNJ = 0.f, sLG = 0.f, sNC = 0.f, sCE = 0.f;
  for (int b = t; b < B_N; b += 256) {
    float nj = (logC - rowLse[b] - jac[b]) * invD;      // nll_joint
    float nc = (0.5f * rowZzy[b] - jac[b]) * invD;      // nll_class
    sNJ += nj;
    sLG += rowSum[b];
    sNC += nc;
    sCE += (nc - nj) * (float)D_N;                      // cat_ce
  }
  __shared__ float r0[256], r1[256], r2[256], r3[256];
  r0[t] = sNJ; r1[t] = sLG; r2[t] = sNC; r3[t] = sCE;
  __syncthreads();
  for (int s = 128; s > 0; s >>= 1) {
    if (t < s) { r0[t] += r0[t + s]; r1[t] += r1[t + s]; r2[t] += r2[t + s]; r3[t] += r3[t + s]; }
    __syncthreads();
  }
  if (t == 0) {
    out[0] = r0[0] / (float)B_N;
    out[1] = r1[0] / ((float)B_N * (float)C_N);
    out[2] = r2[0] / (float)B_N;
    out[3] = r3[0] / (float)B_N;
  }
}

extern "C" void kernel_launch(void* const* d_in, const int* in_sizes, int n_in,
                              void* d_out, int out_size, void* d_ws, size_t ws_size,
                              hipStream_t stream) {
  const float* z   = (const float*)d_in[0];  // (B, D)
  const float* mu  = (const float*)d_in[1];  // (1, C, D)
  const float* jac = (const float*)d_in[2];  // (B,)
  const float* y   = (const float*)d_in[3];  // (B, C)
  float* out = (float*)d_out;

  // Workspace layout
  unsigned short* zh = (unsigned short*)d_ws;                 // B*D bf16
  unsigned short* zl = zh + (size_t)B_N * D_N;                // B*D bf16
  unsigned short* mh = zl + (size_t)B_N * D_N;                // CP*D bf16
  unsigned short* ml = mh + (size_t)CP * D_N;                 // CP*D bf16
  float* z2     = (float*)(ml + (size_t)CP * D_N);            // B
  float* mu2    = z2 + B_N;                                   // CP
  float* zz     = mu2 + CP;                                   // B*CP f32
  float* rowLse = zz + (size_t)B_N * CP;                      // B
  float* rowSum = rowLse + B_N;                               // B
  float* rowZzy = rowSum + B_N;                               // B

  const size_t gemmLds = 2u * 4u * (size_t)TILE * 2u;         // 80 KB dynamic LDS

  k_prep_z<<<B_N, 256, 0, stream>>>(z, zh, zl, z2);
  k_prep_mu<<<CP, 256, 0, stream>>>(mu, mh, ml, mu2);
  k_gemm_zz<<<dim3(CP / 128, B_N / 128), 256, gemmLds, stream>>>(zh, zl, mh, ml, z2, mu2, zz);
  k_row_reduce<<<B_N, 256, 0, stream>>>(zz, y, rowLse, rowSum, rowZzy);
  k_final<<<1, 256, 0, stream>>>(rowLse, rowSum, rowZzy, jac, out);
}
